// Attention_1142461301165
// MI455X (gfx1250) — compile-verified, matrix operand order checked by reading, not run on hardware
//
#include <hip/hip_runtime.h>

#define Bn 16
#define Qn 2048
#define Kn 2048
#define Dn 1024

typedef __attribute__((ext_vector_type(16))) __bf16 v16bf;
typedef __attribute__((ext_vector_type(8)))  __bf16 v8bf;
typedef __attribute__((ext_vector_type(8)))  float  v8f;

// ---------------------------------------------------------------------------
// Fragment loaders for V_WMMA_F32_16X16X32_BF16 (NT layout: both A and B
// operands read 16 rows of a row-major matrix, contraction dim contiguous).
// 16-bit A/B layout (ISA 7.12.2): lane l holds row (l&15); halves:
//   lane<16 : elems 0..7 = K[k0..k0+7],   elems 8..15 = K[k0+16..k0+23]
//   lane>=16: elems 0..7 = K[k0+8..k0+15], elems 8..15 = K[k0+24..k0+31]
// ---------------------------------------------------------------------------
__device__ __forceinline__ v16bf load_frag_bf16(const __bf16* base, int ld,
                                                int row, int k0, int lane) {
    int r  = row + (lane & 15);
    int kb = k0 + ((lane >> 4) << 3);
    const __bf16* p = base + (size_t)r * ld + kb;
    v8bf lo = *(const v8bf*)(p);        // 16B load
    v8bf hi = *(const v8bf*)(p + 16);   // 16B load
    return __builtin_shufflevector(lo, hi, 0, 1, 2, 3, 4, 5, 6, 7,
                                           8, 9, 10, 11, 12, 13, 14, 15);
}

// Same, but source is fp32 (converted to bf16 on the fly).
__device__ __forceinline__ v16bf load_frag_f32cvt(const float* base, int ld,
                                                  int row, int k0, int lane) {
    int r  = row + (lane & 15);
    int kb = k0 + ((lane >> 4) << 3);
    const float* p = base + (size_t)r * ld + kb;
    v8f lo = *(const v8f*)(p);          // 32B load
    v8f hi = *(const v8f*)(p + 16);     // 32B load
    v16bf f;
#pragma unroll
    for (int i = 0; i < 8; ++i) {
        f[i]     = (__bf16)lo[i];
        f[i + 8] = (__bf16)hi[i];
    }
    return f;
}

__device__ __forceinline__ v8f wmma_bf16(v16bf a, v16bf b, v8f c) {
    return __builtin_amdgcn_wmma_f32_16x16x32_bf16(
        false, a, false, b, (short)0, c, false, false);
}

// C/D tile layout (ISA 7.12.2): element i of acc for lane l is
// row = rowBase + ((l>>4)<<3) + i, col = colBase + (l&15)
// ---------------------------------------------------------------------------

// ============================ conversion kernels ===========================
__global__ void cvt_f32_to_bf16(const float* __restrict__ in,
                                __bf16* __restrict__ out, size_t n) {
    size_t i = (size_t)blockIdx.x * blockDim.x + threadIdx.x;
    if (i < n) out[i] = (__bf16)in[i];
}

// context (B,K,D) f32 -> context^T (B,D,K) bf16, LDS-tiled 32x32
__global__ void transpose_cvt(const float* __restrict__ C,
                              __bf16* __restrict__ CT) {
    __shared__ float tile[32][33];
    int b  = blockIdx.z;
    int d0 = blockIdx.x * 32;
    int k0 = blockIdx.y * 32;
    const float* Cb  = C  + (size_t)b * Kn * Dn;
    __bf16*      CTb = CT + (size_t)b * Dn * Kn;
    int tx = threadIdx.x, ty = threadIdx.y;   // (32, 8)
#pragma unroll
    for (int j = 0; j < 32; j += 8)
        tile[ty + j][tx] = Cb[(size_t)(k0 + ty + j) * Dn + d0 + tx];
    __syncthreads();
#pragma unroll
    for (int j = 0; j < 32; j += 8)
        CTb[(size_t)(d0 + ty + j) * Kn + k0 + tx] = (__bf16)tile[tx][ty + j];
}

// ================= GEMM1: scores = Q·C^T, masked, fp32 out =================
// Block = 128 threads (4 waves). Wave tile: 32(M) x 64(N): 8 acc tiles,
// B fragments reused for two row sub-tiles (2 WMMAs per B load).
__global__ void __launch_bounds__(128)
scores_kernel(const __bf16* __restrict__ Qb, const __bf16* __restrict__ Cb,
              const int* __restrict__ mask, float* __restrict__ S) {
    int b    = blockIdx.z;
    int wid  = threadIdx.x >> 5;
    int lane = threadIdx.x & 31;
    int m0 = blockIdx.y * 128 + wid * 32;  // q rows
    int n0 = blockIdx.x * 64;              // k cols
    const __bf16* A  = Qb + (size_t)b * Qn * Dn;
    const __bf16* Bm = Cb + (size_t)b * Kn * Dn;
    float*        Sb = S  + (size_t)b * Qn * Kn;
    const int* mrow  = mask + (size_t)b * Qn;

    v8f acc[2][4] = {};
    for (int d0 = 0; d0 < Dn; d0 += 32) {
        if (d0 + 64 < Dn) {   // global_prefetch of next A tiles
            __builtin_prefetch(A + (size_t)(m0 + (lane & 15)) * Dn + d0 + 64, 0, 1);
            __builtin_prefetch(A + (size_t)(m0 + 16 + (lane & 15)) * Dn + d0 + 64, 0, 1);
        }
        v16bf a0 = load_frag_bf16(A, Dn, m0,      d0, lane);
        v16bf a1 = load_frag_bf16(A, Dn, m0 + 16, d0, lane);
#pragma unroll
        for (int j = 0; j < 4; ++j) {
            v16bf bb = load_frag_bf16(Bm, Dn, n0 + 16 * j, d0, lane);
            acc[0][j] = wmma_bf16(a0, bb, acc[0][j]);
            acc[1][j] = wmma_bf16(a1, bb, acc[1][j]);
        }
    }
    int rb  = m0 + ((lane >> 4) << 3);
    int cc0 = n0 + (lane & 15);
#pragma unroll
    for (int s = 0; s < 2; ++s) {
        bool mk[8];
#pragma unroll
        for (int i = 0; i < 8; ++i) mk[i] = (mrow[rb + 16 * s + i] != 0);
#pragma unroll
        for (int j = 0; j < 4; ++j)
#pragma unroll
            for (int i = 0; i < 8; ++i) {
                float v = mk[i] ? acc[s][j][i] : -__builtin_inff();
                Sb[(size_t)(rb + 16 * s + i) * Kn + cc0 + 16 * j] = v;
            }
    }
}

// ====== softmax over q (axis=1), fused: per-(b,k) online stats + write =====
__global__ void softmax_cols_kernel(float* __restrict__ S) {
    int idx = blockIdx.x * blockDim.x + threadIdx.x;   // b*Kn + k
    if (idx >= Bn * Kn) return;
    int b = idx >> 11;        // / Kn
    int k = idx & (Kn - 1);
    float* p = S + (size_t)b * Qn * Kn + k;
    float m = -__builtin_inff();
    float s = 0.0f;
    for (int q = 0; q < Qn; ++q) {
        float v = p[(size_t)q * Kn];
        if (v > m) {
            s = s * __expf(m - v) + 1.0f;   // exp(-inf)=0 on first hit
            m = v;
        } else if (v > -__builtin_inff()) { // skip masked -inf entries
            s += __expf(v - m);
        }
    }
    float inv = 1.0f / s;
    for (int q = 0; q < Qn; ++q) {
        size_t o = (size_t)q * Kn;
        p[o] = __expf(p[o] - m) * inv;      // exp(-inf - m) = 0 for masked q
    }
}

// ================= GEMM2: mix = A·C  (NT against C^T), bf16 out ============
__global__ void __launch_bounds__(128)
mix_kernel(const float* __restrict__ Attn, const __bf16* __restrict__ CTb,
           __bf16* __restrict__ mixb) {
    int b    = blockIdx.z;
    int wid  = threadIdx.x >> 5;
    int lane = threadIdx.x & 31;
    int m0 = blockIdx.y * 128 + wid * 32;  // q rows
    int n0 = blockIdx.x * 64;              // d cols
    const float*  A  = Attn + (size_t)b * Qn * Kn;
    const __bf16* Bm = CTb  + (size_t)b * Dn * Kn;
    __bf16*       Ob = mixb + (size_t)b * Qn * Dn;

    v8f acc[2][4] = {};
    for (int k0 = 0; k0 < Kn; k0 += 32) {
        if (k0 + 64 < Kn) {
            __builtin_prefetch(A + (size_t)(m0 + (lane & 15)) * Kn + k0 + 64, 0, 1);
            __builtin_prefetch(A + (size_t)(m0 + 16 + (lane & 15)) * Kn + k0 + 64, 0, 1);
        }
        v16bf a0 = load_frag_f32cvt(A, Kn, m0,      k0, lane);
        v16bf a1 = load_frag_f32cvt(A, Kn, m0 + 16, k0, lane);
#pragma unroll
        for (int j = 0; j < 4; ++j) {
            v16bf bb = load_frag_bf16(Bm, Kn, n0 + 16 * j, k0, lane);
            acc[0][j] = wmma_bf16(a0, bb, acc[0][j]);
            acc[1][j] = wmma_bf16(a1, bb, acc[1][j]);
        }
    }
    int rb  = m0 + ((lane >> 4) << 3);
    int cc0 = n0 + (lane & 15);
#pragma unroll
    for (int s = 0; s < 2; ++s)
#pragma unroll
        for (int j = 0; j < 4; ++j)
#pragma unroll
            for (int i = 0; i < 8; ++i)
                Ob[(size_t)(rb + 16 * s + i) * Dn + cc0 + 16 * j] =
                    (__bf16)acc[s][j][i];
}

// ====== GEMM3: out = tanh([mix | query] · w_out^T), two-phase accumulate ===
__global__ void __launch_bounds__(128)
out_kernel(const __bf16* __restrict__ mixb, const __bf16* __restrict__ Qb,
           const __bf16* __restrict__ Wb, float* __restrict__ Out) {
    int b    = blockIdx.z;
    int wid  = threadIdx.x >> 5;
    int lane = threadIdx.x & 31;
    int m0 = blockIdx.y * 128 + wid * 32;  // q rows
    int n0 = blockIdx.x * 64;              // output d cols
    const __bf16* A1 = mixb + (size_t)b * Qn * Dn;
    const __bf16* A2 = Qb   + (size_t)b * Qn * Dn;
    float*        Ob = Out  + (size_t)b * Qn * Dn;
    const int ldw = 2 * Dn;                // w_out row n: e = 0..2D-1

    v8f acc[2][4] = {};
    for (int e0 = 0; e0 < Dn; e0 += 32) {          // mix part (e < D)
        v16bf a0 = load_frag_bf16(A1, Dn, m0,      e0, lane);
        v16bf a1 = load_frag_bf16(A1, Dn, m0 + 16, e0, lane);
#pragma unroll
        for (int j = 0; j < 4; ++j) {
            v16bf bb = load_frag_bf16(Wb, ldw, n0 + 16 * j, e0, lane);
            acc[0][j] = wmma_bf16(a0, bb, acc[0][j]);
            acc[1][j] = wmma_bf16(a1, bb, acc[1][j]);
        }
    }
    for (int e0 = 0; e0 < Dn; e0 += 32) {          // query part (e >= D)
        v16bf a0 = load_frag_bf16(A2, Dn, m0,      e0, lane);
        v16bf a1 = load_frag_bf16(A2, Dn, m0 + 16, e0, lane);
#pragma unroll
        for (int j = 0; j < 4; ++j) {
            v16bf bb = load_frag_bf16(Wb, ldw, n0 + 16 * j, Dn + e0, lane);
            acc[0][j] = wmma_bf16(a0, bb, acc[0][j]);
            acc[1][j] = wmma_bf16(a1, bb, acc[1][j]);
        }
    }
    int rb  = m0 + ((lane >> 4) << 3);
    int cc0 = n0 + (lane & 15);
#pragma unroll
    for (int s = 0; s < 2; ++s)
#pragma unroll
        for (int j = 0; j < 4; ++j)
#pragma unroll
            for (int i = 0; i < 8; ++i)
                Ob[(size_t)(rb + 16 * s + i) * Dn + cc0 + 16 * j] =
                    tanhf(acc[s][j][i]);
}

// ===========================================================================
extern "C" void kernel_launch(void* const* d_in, const int* in_sizes, int n_in,
                              void* d_out, int out_size, void* d_ws, size_t ws_size,
                              hipStream_t stream) {
    const float* query   = (const float*)d_in[0];   // (B,Q,D) f32
    const float* context = (const float*)d_in[1];   // (B,K,D) f32
    const int*   amask   = (const int*)  d_in[2];   // (B,Q)   i32
    const float* w_out   = (const float*)d_in[3];   // (D,2D)  f32

    float* out_f = (float*)d_out;                               // (B,Q,D)
    float* attn  = (float*)d_out + (size_t)Bn * Qn * Dn;        // (B,Q,K)

    // workspace layout
    char* ws = (char*)d_ws;
    const size_t nQD = (size_t)Bn * Qn * Dn;     // 33 554 432
    const size_t nKD = (size_t)Bn * Kn * Dn;     // 33 554 432
    const size_t nW  = (size_t)Dn * 2 * Dn;      //  2 097 152
    __bf16* qbf   = (__bf16*)(ws);                              // 64 MiB
    __bf16* cbf   = (__bf16*)(ws + nQD * 2);                    // 64 MiB
    __bf16* ctbf  = (__bf16*)(ws + nQD * 2 + nKD * 2);          // 64 MiB
    __bf16* wbf   = (__bf16*)(ws + nQD * 2 + nKD * 2 * 2);      //  4 MiB
    __bf16* mixbf = (__bf16*)(ws + nQD * 2 + nKD * 2 * 2 + nW * 2);  // 64 MiB

    // 1) precision conversion + transposed context
    cvt_f32_to_bf16<<<(nQD + 255) / 256, 256, 0, stream>>>(query, qbf, nQD);
    cvt_f32_to_bf16<<<(nKD + 255) / 256, 256, 0, stream>>>(context, cbf, nKD);
    cvt_f32_to_bf16<<<(nW + 255) / 256, 256, 0, stream>>>(w_out, wbf, nW);
    transpose_cvt<<<dim3(Dn / 32, Kn / 32, Bn), dim3(32, 8), 0, stream>>>(context, ctbf);

    // 2) scores = Q·C^T with q-row masking (written into attn region of d_out)
    scores_kernel<<<dim3(Kn / 64, Qn / 128, Bn), 128, 0, stream>>>(qbf, cbf, amask, attn);

    // 3) softmax over axis=1 (per (b,k) column), fused stats + normalize
    softmax_cols_kernel<<<(Bn * Kn + 255) / 256, 256, 0, stream>>>(attn);

    // 4) mix = A·C   (A fp32 converted on the fly, B = C^T bf16)
    mix_kernel<<<dim3(Dn / 64, Qn / 128, Bn), 128, 0, stream>>>(attn, ctbf, mixbf);

    // 5) out = tanh([mix | query] · w_out^T)
    out_kernel<<<dim3(Dn / 64, Qn / 128, Bn), 128, 0, stream>>>(mixbf, qbf, wbf, out_f);
}
